// Speller_58574763983102
// MI455X (gfx1250) — matrix operand order repaired and checked by reading, not compiled, not run on hardware
//
#include <hip/hip_runtime.h>
#include <hip/hip_bf16.h>
#include <stdint.h>

// Problem constants (match reference)
#define BB 32
#define TT 512
#define HH 1024
#define SS 64
#define VV 10000

typedef _Float16 half16 __attribute__((ext_vector_type(16)));
typedef float    f32x8  __attribute__((ext_vector_type(8)));

union AF { half16 h; uint32_t u[8]; };

// ---------------- WMMA fragment loaders ----------------
// 16-bit A-matrix 16x32 (MxK) layout (ISA 7.12.2):
// lane L: M = L%16, half = L/16; VGPR v holds K = (v/4)*16 + half*8 + (v%4)*2, +1
__device__ __forceinline__ void load_a_frag(const _Float16* __restrict__ A, int K,
                                            int m, int half, int kc, AF& f) {
#pragma unroll
  for (int v = 0; v < 8; ++v) {
    int k = kc + ((v >> 2) << 4) + (half << 3) + ((v & 3) << 1);
    f.u[v] = *(const uint32_t*)(A + (size_t)m * K + k);
  }
}

// B-matrix 32x16 (KxN): lane L: N = L%16, half = L/16; VGPR v holds K = (v/4)*16 + half*8 + (v%4)*2, +1
template <typename WT>
__device__ __forceinline__ void load_w_frag(const WT* __restrict__ W, int K,
                                            int n, int khalf, int kc, AF& f) {
  const size_t base = (size_t)n * K + kc + (khalf << 3);
#pragma unroll
  for (int v = 0; v < 8; ++v) {
    if constexpr (sizeof(WT) == 2) {
      f.u[v] = *(const uint32_t*)((const _Float16*)W + base + ((v >> 2) << 4) + ((v & 3) << 1));
    } else {
      const float* p = (const float*)W + base + ((v >> 2) << 4) + ((v & 3) << 1);
      f.h[2 * v]     = (_Float16)p[0];
      f.h[2 * v + 1] = (_Float16)p[1];
    }
  }
}

// ---------------- GEMM: C[32,N] = A0[32,K0]@W0[N,K0]^T (+ A1[32,K1]@W1[N,K1]^T) + bias ----
// Each wave owns TWO 16-wide N tiles (A-fragment reuse: 4 WMMA per 4 fragment loads),
// with register double-buffering so next-chunk loads overlap current WMMAs.
template <typename WT, int K0, int K1>
__global__ void k_gemm(const _Float16* __restrict__ A0,
                       const _Float16* __restrict__ A1,
                       const WT* __restrict__ W0, const WT* __restrict__ W1,
                       const float* __restrict__ bias,
                       float* __restrict__ C, int N, int Ntiles) {
  const int gw   = blockIdx.x * (blockDim.x >> 5) + (threadIdx.x >> 5);
  const int lane = threadIdx.x & 31;
  const int nt0  = gw * 2;
  if (nt0 >= Ntiles) return;
  const bool has1 = (nt0 + 1) < Ntiles;
  const int nt1   = has1 ? (nt0 + 1) : nt0;  // clamp: loads stay in-bounds, result unused

  const int nlo = lane & 15, khalf = lane >> 4;
  const int n0 = nt0 * 16 + nlo;
  const int n1 = nt1 * 16 + nlo;
  const int m0 = nlo, m1 = 16 + nlo;

  f32x8 c00 = {}, c01 = {}, c10 = {}, c11 = {};

  AF b0, b1, a0, a1, nb0, nb1, na0, na1;

  auto load_chunk = [&](const _Float16* __restrict__ A, const WT* __restrict__ W, int K_,
                        int kc, AF& fb0, AF& fb1, AF& fa0, AF& fa1) {
    load_w_frag<WT>(W, K_, n0, khalf, kc, fb0);
    load_w_frag<WT>(W, K_, n1, khalf, kc, fb1);
    load_a_frag(A, K_, m0, khalf, kc, fa0);
    load_a_frag(A, K_, m1, khalf, kc, fa1);
  };
  auto mma4 = [&](AF& fb0, AF& fb1, AF& fa0, AF& fa1) {
    c00 = __builtin_amdgcn_wmma_f32_16x16x32_f16(false, fa0.h, false, fb0.h, (short)0, c00, false, false);
    c10 = __builtin_amdgcn_wmma_f32_16x16x32_f16(false, fa1.h, false, fb0.h, (short)0, c10, false, false);
    c01 = __builtin_amdgcn_wmma_f32_16x16x32_f16(false, fa0.h, false, fb1.h, (short)0, c01, false, false);
    c11 = __builtin_amdgcn_wmma_f32_16x16x32_f16(false, fa1.h, false, fb1.h, (short)0, c11, false, false);
  };

  load_chunk(A0, W0, K0, 0, b0, b1, a0, a1);
#pragma unroll 4
  for (int kc = 32; kc < K0; kc += 32) {
    __builtin_prefetch((const void*)(W0 + (size_t)n0 * K0 + kc + 32), 0, 1);
    load_chunk(A0, W0, K0, kc, nb0, nb1, na0, na1);
    mma4(b0, b1, a0, a1);
    b0 = nb0; b1 = nb1; a0 = na0; a1 = na1;
  }
  if constexpr (K1 > 0) {
    load_chunk(A1, W1, K1, 0, nb0, nb1, na0, na1);
    mma4(b0, b1, a0, a1);
    b0 = nb0; b1 = nb1; a0 = na0; a1 = na1;
#pragma unroll 4
    for (int kc = 32; kc < K1; kc += 32) {
      __builtin_prefetch((const void*)(W1 + (size_t)n0 * K1 + kc + 32), 0, 1);
      load_chunk(A1, W1, K1, kc, nb0, nb1, na0, na1);
      mma4(b0, b1, a0, a1);
      b0 = nb0; b1 = nb1; a0 = na0; a1 = na1;
    }
  }
  mma4(b0, b1, a0, a1);

  // C/D layout: VGPR r, lanes 0-15: M=r; lanes 16-31: M=r+8
  const float bv0 = bias[n0];
  const float bv1 = bias[n1];
#pragma unroll
  for (int r = 0; r < 8; ++r) {
    int mA = r + 8 * khalf;
    int mB = 16 + r + 8 * khalf;
    C[(size_t)mA * N + n0] = c00[r] + bv0;
    C[(size_t)mB * N + n0] = c10[r] + bv0;
    if (has1) {
      C[(size_t)mA * N + n1] = c01[r] + bv1;
      C[(size_t)mB * N + n1] = c11[r] + bv1;
    }
  }
}

// ---------------- helpers ----------------
__global__ void k_minlen(const int* __restrict__ enc, int* __restrict__ out) {
  if (threadIdx.x == 0) {
    int m = 0x7fffffff;
    for (int i = 0; i < BB; ++i) m = enc[i] < m ? enc[i] : m;
    *out = m;
  }
}

__global__ void k_cvt_f16(const float* __restrict__ src, _Float16* __restrict__ dst, size_t n) {
  size_t i = (size_t)blockIdx.x * blockDim.x + threadIdx.x;
  if (i < n) dst[i] = (_Float16)src[i];
}

__global__ void k_bias_sum(const float* __restrict__ a, const float* __restrict__ b,
                           const float* __restrict__ c, const float* __restrict__ d,
                           float* __restrict__ o1, float* __restrict__ o2) {
  int i = blockIdx.x * 256 + threadIdx.x;  // 4096
  o1[i] = a[i] + b[i];
  o2[i] = c[i] + d[i];
}

__global__ void k_init_state(const float* __restrict__ s0, const float* __restrict__ c10,
                             const float* __restrict__ c20, const float* __restrict__ s20,
                             float* __restrict__ s, _Float16* __restrict__ s16,
                             float* __restrict__ c1, float* __restrict__ c2,
                             float* __restrict__ s2, _Float16* __restrict__ s2_16) {
  int i = blockIdx.x * 256 + threadIdx.x;  // 32768
  float a = s0[i], d = s20[i];
  s[i] = a;   s16[i]  = (_Float16)a;
  c1[i] = c10[i];
  c2[i] = c20[i];
  s2[i] = d;  s2_16[i] = (_Float16)d;
}

// xcat[32,2048] = [ embed[y[:,step]] | ctx ] as f16
__global__ void k_build_xcat(const float* __restrict__ embed, const int* __restrict__ y,
                             const float* __restrict__ ctx, int step,
                             _Float16* __restrict__ xcat) {
  int i = blockIdx.x * 256 + threadIdx.x;  // 65536
  int b = i >> 11, col = i & 2047;
  float v;
  if (col < HH) {
    int tok = y[b * SS + step];
    v = embed[(size_t)tok * HH + col];
  } else {
    v = ctx[b * HH + (col - HH)];
  }
  xcat[i] = (_Float16)v;
}

// z[32,2048] = [ s2 | ctx ] as f16
__global__ void k_build_z(const float* __restrict__ s2, const float* __restrict__ ctx,
                          _Float16* __restrict__ z) {
  int i = blockIdx.x * 256 + threadIdx.x;  // 65536
  int b = i >> 11, col = i & 2047;
  float v = (col < HH) ? s2[b * HH + col] : ctx[b * HH + (col - HH)];
  z[i] = (_Float16)v;
}

// LSTM pointwise: gate order i,f,g,o; updates c in place, writes h (f32 + f16)
__global__ void k_lstm_point(const float* __restrict__ g, float* __restrict__ c,
                             float* __restrict__ h, _Float16* __restrict__ h16) {
  int i = blockIdx.x * 256 + threadIdx.x;  // 32768
  int b = i >> 10, col = i & 1023;
  const float* gb = g + (size_t)b * (4 * HH);
  float gi = gb[col], gf = gb[HH + col], gg = gb[2 * HH + col], go = gb[3 * HH + col];
  float si = 1.f / (1.f + expf(-gi));
  float sf = 1.f / (1.f + expf(-gf));
  float so = 1.f / (1.f + expf(-go));
  float cn = sf * c[i] + si * tanhf(gg);
  float hn = so * tanhf(cn);
  c[i] = cn;
  h[i] = hn;
  h16[i] = (_Float16)hn;
}

// attention: one block per batch row; scores -> softmax -> ctx
__global__ void k_attn(const float* __restrict__ hidden, const float* __restrict__ s,
                       const int* __restrict__ minlen, float* __restrict__ ctx) {
  __shared__ float sc[TT];
  __shared__ float red[256];
  __shared__ float ssh[HH];
  const int b = blockIdx.x, tid = threadIdx.x;
  const bool msk = (b >= *minlen);
  const float* hb = hidden + (size_t)b * TT * HH;
  const float* sb = s + b * HH;
  for (int hcol = tid; hcol < HH; hcol += 256) ssh[hcol] = sb[hcol];
  __syncthreads();
  for (int t = tid; t < TT; t += 256) {
    const float* row = hb + (size_t)t * HH;
    float acc = 0.f;
    for (int hc = 0; hc < HH; ++hc) {
      float hv = msk ? -1e9f : row[hc];
      acc += hv * ssh[hc];
    }
    sc[t] = acc;
  }
  __syncthreads();
  // softmax over T=512
  float m = fmaxf(sc[tid], sc[tid + 256]);
  red[tid] = m; __syncthreads();
  for (int off = 128; off > 0; off >>= 1) {
    if (tid < off) red[tid] = fmaxf(red[tid], red[tid + off]);
    __syncthreads();
  }
  const float mx = red[0]; __syncthreads();
  float e0 = expf(sc[tid] - mx), e1 = expf(sc[tid + 256] - mx);
  sc[tid] = e0; sc[tid + 256] = e1;
  red[tid] = e0 + e1; __syncthreads();
  for (int off = 128; off > 0; off >>= 1) {
    if (tid < off) red[tid] += red[tid + off];
    __syncthreads();
  }
  const float inv = 1.f / red[0]; __syncthreads();
  sc[tid] *= inv; sc[tid + 256] *= inv;
  __syncthreads();
  for (int hc = tid; hc < HH; hc += 256) {
    float acc = 0.f;
    for (int t = 0; t < TT; ++t) {
      float hv = msk ? -1e9f : hb[(size_t)t * HH + hc];
      acc += sc[t] * hv;
    }
    ctx[b * HH + hc] = acc;
  }
}

// log_softmax over V and scatter into out[b, step, :]
__global__ void k_logsoftmax(const float* __restrict__ logits, float* __restrict__ out, int step) {
  __shared__ float red[256];
  const int b = blockIdx.x, tid = threadIdx.x;
  const float* lb = logits + (size_t)b * VV;
  float m = -3.4e38f;
  for (int v = tid; v < VV; v += 256) m = fmaxf(m, lb[v]);
  red[tid] = m; __syncthreads();
  for (int off = 128; off > 0; off >>= 1) {
    if (tid < off) red[tid] = fmaxf(red[tid], red[tid + off]);
    __syncthreads();
  }
  const float mx = red[0]; __syncthreads();
  float s = 0.f;
  for (int v = tid; v < VV; v += 256) s += expf(lb[v] - mx);
  red[tid] = s; __syncthreads();
  for (int off = 128; off > 0; off >>= 1) {
    if (tid < off) red[tid] += red[tid + off];
    __syncthreads();
  }
  const float lse = mx + logf(red[0]);
  float* ob = out + ((size_t)b * SS + step) * VV;
  for (int v = tid; v < VV; v += 256) ob[v] = lb[v] - lse;
}

// ---------------- host ----------------
template <int K0, int K1>
static void launch_gemm(bool useF16,
                        const _Float16* A0, const _Float16* A1,
                        const _Float16* Wa16, const float* Wa32,
                        const _Float16* Wb16, const float* Wb32,
                        const float* bias, float* C, int N, int NT, hipStream_t stream) {
  const int waves  = (NT + 1) / 2;   // 2 N-tiles per wave
  const int blocks = (waves + 1) / 2;  // 2 waves per block
  if (useF16)
    k_gemm<_Float16, K0, K1><<<blocks, 64, 0, stream>>>(A0, A1, Wa16, Wb16, bias, C, N, NT);
  else
    k_gemm<float, K0, K1><<<blocks, 64, 0, stream>>>(A0, A1, Wa32, Wb32, bias, C, N, NT);
}

extern "C" void kernel_launch(void* const* d_in, const int* in_sizes, int n_in,
                              void* d_out, int out_size, void* d_ws, size_t ws_size,
                              hipStream_t stream) {
  (void)in_sizes; (void)n_in; (void)out_size;
  const float* hidden = (const float*)d_in[0];
  const int*   y      = (const int*)d_in[1];
  const int*   enc    = (const int*)d_in[2];
  const float* embed  = (const float*)d_in[3];
  const float* Wih1   = (const float*)d_in[4];
  const float* Whh1   = (const float*)d_in[5];
  const float* bih1   = (const float*)d_in[6];
  const float* bhh1   = (const float*)d_in[7];
  const float* Wih2   = (const float*)d_in[8];
  const float* Whh2   = (const float*)d_in[9];
  const float* bih2   = (const float*)d_in[10];
  const float* bhh2   = (const float*)d_in[11];
  const float* Wmlp   = (const float*)d_in[12];
  const float* bmlp   = (const float*)d_in[13];
  const float* s0     = (const float*)d_in[14];
  const float* c10    = (const float*)d_in[15];
  const float* c20    = (const float*)d_in[16];
  const float* s20    = (const float*)d_in[17];
  float* out = (float*)d_out;

  char* ws = (char*)d_ws;
  size_t off = 0;
  auto alloc = [&](size_t bytes) -> void* {
    void* p = ws + off;
    off += (bytes + 255) & ~(size_t)255;
    return p;
  };

  int*       minlen = (int*)alloc(256);
  float*     bsum1  = (float*)alloc(4096 * 4);
  float*     bsum2  = (float*)alloc(4096 * 4);
  float*     sbuf   = (float*)alloc(BB * HH * 4);
  _Float16*  s16    = (_Float16*)alloc(BB * HH * 2);
  float*     s2buf  = (float*)alloc(BB * HH * 4);
  _Float16*  s2_16  = (_Float16*)alloc(BB * HH * 2);
  float*     c1     = (float*)alloc(BB * HH * 4);
  float*     c2     = (float*)alloc(BB * HH * 4);
  float*     ctx    = (float*)alloc(BB * HH * 4);
  _Float16*  xcat   = (_Float16*)alloc(BB * 2 * HH * 2);
  _Float16*  zbuf   = (_Float16*)alloc(BB * 2 * HH * 2);
  float*     gates  = (float*)alloc(BB * 4 * HH * 4);
  float*     logits = (float*)alloc((size_t)BB * VV * 4);

  const size_t nIh1 = (size_t)4 * HH * 2 * HH;  // 4096x2048
  const size_t nHh1 = (size_t)4 * HH * HH;      // 4096x1024
  const size_t nIh2 = (size_t)4 * HH * HH;
  const size_t nHh2 = (size_t)4 * HH * HH;
  const size_t nMlp = (size_t)VV * 2 * HH;      // 10000x2048
  const size_t needW16 = 2 * (nIh1 + nHh1 + nIh2 + nHh2 + nMlp) + 8 * 256;
  const bool useF16 = (ws_size >= off + needW16);

  const _Float16 *w16ih1 = nullptr, *w16hh1 = nullptr, *w16ih2 = nullptr,
                 *w16hh2 = nullptr, *w16mlp = nullptr;

  // ---- one-time (per launch) prep ----
  k_minlen<<<1, 32, 0, stream>>>(enc, minlen);
  k_bias_sum<<<16, 256, 0, stream>>>(bih1, bhh1, bih2, bhh2, bsum1, bsum2);
  k_init_state<<<128, 256, 0, stream>>>(s0, c10, c20, s20, sbuf, s16, c1, c2, s2buf, s2_16);

  if (useF16) {
    auto cvt = [&](const float* src, size_t n) -> _Float16* {
      _Float16* p = (_Float16*)alloc(n * 2);
      k_cvt_f16<<<(unsigned)((n + 255) / 256), 256, 0, stream>>>(src, p, n);
      return p;
    };
    w16ih1 = cvt(Wih1, nIh1);
    w16hh1 = cvt(Whh1, nHh1);
    w16ih2 = cvt(Wih2, nIh2);
    w16hh2 = cvt(Whh2, nHh2);
    w16mlp = cvt(Wmlp, nMlp);
  }

  // ctx0 = attn(s0)
  k_attn<<<BB, 256, 0, stream>>>(hidden, sbuf, minlen, ctx);

  for (int t = 0; t < SS; ++t) {
    k_build_xcat<<<256, 256, 0, stream>>>(embed, y, ctx, t, xcat);
    // LSTM1 gates: xcat @ Wih1^T + s_prev @ Whh1^T + (bih1+bhh1)
    launch_gemm<2 * HH, HH>(useF16, xcat, s16, w16ih1, Wih1, w16hh1, Whh1,
                            bsum1, gates, 4 * HH, 256, stream);
    k_lstm_point<<<128, 256, 0, stream>>>(gates, c1, sbuf, s16);
    // LSTM2 gates: s @ Wih2^T + s2_prev @ Whh2^T + (bih2+bhh2)
    launch_gemm<HH, HH>(useF16, s16, s2_16, w16ih2, Wih2, w16hh2, Whh2,
                        bsum2, gates, 4 * HH, 256, stream);
    k_lstm_point<<<128, 256, 0, stream>>>(gates, c2, s2buf, s2_16);
    // attention on s2
    k_attn<<<BB, 256, 0, stream>>>(hidden, s2buf, minlen, ctx);
    // logits = [s2|ctx] @ Wmlp^T + bmlp
    k_build_z<<<256, 256, 0, stream>>>(s2buf, ctx, zbuf);
    launch_gemm<2 * HH, 0>(useF16, zbuf, nullptr, w16mlp, Wmlp, nullptr, nullptr,
                           bmlp, logits, VV, VV / 16, stream);
    k_logsoftmax<<<BB, 256, 0, stream>>>(logits, out, t);
  }
}
